// MaskedAttentionLayer_609885356240
// MI455X (gfx1250) — matrix operand order repaired and checked by reading, not compiled, not run on hardware
//
#include <hip/hip_runtime.h>
#include <hip/hip_bf16.h>
#include <math.h>

// Problem dims from the reference
#define BATCH 4
#define SEQ   4096
#define DIM   256
#define NT16  (SEQ / 16)

typedef __attribute__((ext_vector_type(16))) __bf16 v16bf;
typedef __attribute__((ext_vector_type(8)))  float  v8f;

// 32-byte bf16 fragment viewed as two 16-byte loads / raw shorts
union ABFrag {
    v16bf v;
    uint4 u[2];
    unsigned short s[16];
};

__device__ __forceinline__ unsigned short f2bf(float f) {
    unsigned int u = __float_as_uint(f);
    unsigned int r = u + 0x7FFFu + ((u >> 16) & 1u);   // round-to-nearest-even
    return (unsigned short)(r >> 16);
}

__device__ __forceinline__ void load_frag(ABFrag& f, const unsigned short* p) {
    f.u[0] = *(const uint4*)(p);
    f.u[1] = *(const uint4*)(p + 16);
}

// ---------------------------------------------------------------------------
// Kernel 1: fp32 -> bf16 conversion (x and the three weight matrices)
// ---------------------------------------------------------------------------
__global__ void attn_cvt_bf16(const float* __restrict__ src,
                              unsigned short* __restrict__ dst, int n) {
    int i = blockIdx.x * blockDim.x + threadIdx.x;
    int stride = gridDim.x * blockDim.x;
    for (; i < n; i += stride) dst[i] = f2bf(src[i]);
}

// ---------------------------------------------------------------------------
// Kernel 2: QKV projection.  q = x @ Wq^T etc.
//   A = x rows (bf16); B columns = rows of W (W row-major is already the
//   per-lane-contiguous layout WMMA's B fragment wants).
//   Q, K stored [B][N][D]; V stored transposed [B][D][N].
//   3 waves per block: wave 0->Q, 1->K, 2->V(T). One block per 16-row tile.
// ---------------------------------------------------------------------------
__global__ void attn_qkv_gemm(const unsigned short* __restrict__ xb,
                              const unsigned short* __restrict__ wq,
                              const unsigned short* __restrict__ wk,
                              const unsigned short* __restrict__ wv,
                              unsigned short* __restrict__ Qb,
                              unsigned short* __restrict__ Kb,
                              unsigned short* __restrict__ Vtb) {
    const int wave = threadIdx.x >> 5;       // 0:Q 1:K 2:V
    const int lane = threadIdx.x & 31;
    const int b    = blockIdx.x / NT16;
    const int n0   = (blockIdx.x % NT16) * 16;

    // wave-uniform operand / destination selection (computed once, branch-free
    // store loop afterwards)
    const unsigned short* W = (wave == 0) ? wq : (wave == 1) ? wk : wv;
    unsigned short* dst;
    size_t rstride, cstride;
    if (wave == 0)      { dst = Qb  + ((size_t)b * SEQ + n0) * DIM; rstride = DIM; cstride = 1; }
    else if (wave == 1) { dst = Kb  + ((size_t)b * SEQ + n0) * DIM; rstride = DIM; cstride = 1; }
    else                { dst = Vtb + (size_t)b * DIM * SEQ + n0;   rstride = 1;   cstride = SEQ; }

    const int rowA = lane & 15;
    const int aoff = (lane < 16) ? 0 : 8;    // A frag: K 0..7/16..23 vs 8..15/24..31
    const int colB = lane & 15;
    const int boff = (lane < 16) ? 0 : 8;    // B frag: K 0..7/16..23 vs 8..15/24..31
    const int colC = lane & 15;
    const int rowCB = (lane >> 4) << 3;

    // Cache the 16x256 x tile as 8 A fragments (K = 32 per fragment)
    ABFrag a[8];
    const unsigned short* xrow = xb + ((size_t)b * SEQ + n0 + rowA) * DIM;
#pragma unroll
    for (int e = 0; e < 8; ++e)
        load_frag(a[e], xrow + e * 32 + aoff);

    for (int dt = 0; dt < 16; ++dt) {
        v8f acc = {};
        const unsigned short* wcol = W + (size_t)(dt * 16 + colB) * DIM;
        // double-buffered B fragments: overlap next loads with current WMMA
        ABFrag bf[2];
        load_frag(bf[0], wcol + boff);
#pragma unroll
        for (int e = 0; e < 8; ++e) {
            if (e + 1 < 8)
                load_frag(bf[(e + 1) & 1], wcol + (e + 1) * 32 + boff);
            acc = __builtin_amdgcn_wmma_f32_16x16x32_bf16(
                false, a[e].v, false, bf[e & 1].v, (short)0, acc, false, false);
        }
#pragma unroll
        for (int r = 0; r < 8; ++r) {
            dst[(size_t)(rowCB + r) * rstride + (size_t)(dt * 16 + colC) * cstride] =
                f2bf(acc[r]);
        }
    }
}

// ---------------------------------------------------------------------------
// Kernel 3: per-COLUMN softmax stats (reference softmax is over axis=1 = n).
//   One wave owns 16 columns m, streams all n tiles, online (max, sum-exp).
//   Mask quirk: position (n,m) masked when m <= n (keeps the future).
// ---------------------------------------------------------------------------
__global__ void attn_col_stats(const unsigned short* __restrict__ Qb,
                               const unsigned short* __restrict__ Kb,
                               float* __restrict__ colmax,
                               float* __restrict__ colsum) {
    const int lane = threadIdx.x & 31;
    const int b    = blockIdx.x / NT16;
    const int m0   = (blockIdx.x % NT16) * 16;
    const float scale = 0.0625f;             // 1/sqrt(256)

    const int rowA = lane & 15;
    const int aoff = (lane < 16) ? 0 : 8;
    const int colB = lane & 15;
    const int boff = (lane < 16) ? 0 : 8;
    const int colC = lane & 15;
    const int rowCB = (lane >> 4) << 3;
    const int m = m0 + colC;

    // Cache B fragments: columns m0..m0+15 of S-GEMM B  ==  rows of K
    ABFrag bt[8];
    const unsigned short* krow = Kb + ((size_t)b * SEQ + m0 + colB) * DIM;
#pragma unroll
    for (int e = 0; e < 8; ++e)
        load_frag(bt[e], krow + e * 32 + boff);

    const unsigned short* qbase = Qb + ((size_t)b * SEQ + rowA) * DIM;

    float mx = -3.0e38f, sum = 0.0f;
    for (int nt = 0; nt < NT16; ++nt) {
        v8f acc = {};
        const unsigned short* qrow = qbase + (size_t)nt * 16 * DIM;
        // double-buffered A fragments streamed from Q
        ABFrag a[2];
        load_frag(a[0], qrow + aoff);
#pragma unroll
        for (int e = 0; e < 8; ++e) {
            if (e + 1 < 8)
                load_frag(a[(e + 1) & 1], qrow + (e + 1) * 32 + aoff);
            acc = __builtin_amdgcn_wmma_f32_16x16x32_bf16(
                false, a[e & 1].v, false, bt[e].v, (short)0, acc, false, false);
        }
#pragma unroll
        for (int r = 0; r < 8; ++r) {
            const int n = nt * 16 + rowCB + r;
            const float s = (acc[r] - ((m <= n) ? 1.0e10f : 0.0f)) * scale;
            if (s > mx) { sum = sum * __expf(mx - s) + 1.0f; mx = s; }
            else        { sum += __expf(s - mx); }
        }
    }
    // lanes L and L^16 hold the same column; merge their partial stats
    const float mx2  = __shfl_xor(mx, 16, 32);
    const float sum2 = __shfl_xor(sum, 16, 32);
    const float M = fmaxf(mx, mx2);
    const float S = sum * __expf(mx - M) + sum2 * __expf(mx2 - M);
    if (lane < 16) {
        colmax[b * SEQ + m] = M;
        colsum[b * SEQ + m] = S;
    }
}

// ---------------------------------------------------------------------------
// Kernel 4: out = P @ V with S recomputed on the fly (L2-resident operands).
//   One wave per 16-row output tile; 16 fp32 C accumulators cover all 256 d.
//   Per 32-key step: 16 WMMA for the score tiles, C->A relayout through a
//   1 KB LDS staging buffer (exp + normalize fused), 16 WMMA against V^T.
// ---------------------------------------------------------------------------
__global__ void attn_out_gemm(const unsigned short* __restrict__ Qb,
                              const unsigned short* __restrict__ Kb,
                              const unsigned short* __restrict__ Vtb,
                              const float* __restrict__ colmax,
                              const float* __restrict__ colsum,
                              float* __restrict__ out) {
    __shared__ alignas(16) unsigned short Pbuf[16 * 32];

    const int lane = threadIdx.x & 31;
    const int b    = blockIdx.x / NT16;
    const int n0   = (blockIdx.x % NT16) * 16;
    const float scale = 0.0625f;

    const int rowA = lane & 15;
    const int aoff = (lane < 16) ? 0 : 8;
    const int colB = lane & 15;
    const int boff = (lane < 16) ? 0 : 8;
    const int colC = lane & 15;
    const int rowCB = (lane >> 4) << 3;

    // Cache Q A-fragments for this row tile (reused for all 128 m-steps)
    ABFrag aQ[8];
    const unsigned short* qrow = Qb + ((size_t)b * SEQ + n0 + rowA) * DIM;
#pragma unroll
    for (int e = 0; e < 8; ++e)
        load_frag(aQ[e], qrow + e * 32 + aoff);

    const unsigned short* kbase = Kb + ((size_t)b * SEQ + colB) * DIM;
    const unsigned short* vbase = Vtb + ((size_t)b * DIM + colB) * SEQ;

    v8f acc[16] = {};

    for (int ms = 0; ms < SEQ / 32; ++ms) {
        const int m0 = ms * 32;

        // column stats: lane c holds stats for column m0+c
        const float cmAll = colmax[b * SEQ + m0 + lane];
        const float csAll = colsum[b * SEQ + m0 + lane];

        // gfx1250 prefetch of the next K tile
        if (ms + 1 < SEQ / 32)
            __builtin_prefetch(kbase + (size_t)(m0 + 32) * DIM, 0, 1);

        // Recompute two 16x16 score tiles, convert to P (bf16) into LDS
#pragma unroll
        for (int half = 0; half < 2; ++half) {
            v8f sacc = {};
            const unsigned short* krow = kbase + (size_t)(m0 + half * 16) * DIM;
            // double-buffered K fragments
            ABFrag bk[2];
            load_frag(bk[0], krow + boff);
#pragma unroll
            for (int e = 0; e < 8; ++e) {
                if (e + 1 < 8)
                    load_frag(bk[(e + 1) & 1], krow + (e + 1) * 32 + boff);
                sacc = __builtin_amdgcn_wmma_f32_16x16x32_bf16(
                    false, aQ[e].v, false, bk[e & 1].v, (short)0, sacc, false, false);
            }
            const int mloc = half * 16 + colC;
            const int m    = m0 + mloc;
            const float cm  = __shfl(cmAll, mloc, 32);
            const float inv = 1.0f / __shfl(csAll, mloc, 32);
#pragma unroll
            for (int r = 0; r < 8; ++r) {
                const int n = n0 + rowCB + r;
                const float s = (sacc[r] - ((m <= n) ? 1.0e10f : 0.0f)) * scale;
                const float p = __expf(s - cm) * inv;
                Pbuf[(rowCB + r) * 32 + mloc] = f2bf(p);
            }
        }
        __syncthreads();

        // Reload P as an A fragment (16x32, per-lane contiguous K runs)
        ABFrag aP;
        {
            const unsigned short* prow = &Pbuf[rowA * 32];
            aP.u[0] = *(const uint4*)(prow + aoff);
            aP.u[1] = *(const uint4*)(prow + aoff + 16);
        }

        // P @ V: 16 d-tiles, B columns = rows of V^T (contiguous along m);
        // double-buffered V fragments
        {
            ABFrag bv[2];
            load_frag(bv[0], vbase + m0 + boff);
#pragma unroll
            for (int dt = 0; dt < 16; ++dt) {
                if (dt + 1 < 16)
                    load_frag(bv[(dt + 1) & 1],
                              vbase + (size_t)(dt + 1) * 16 * SEQ + m0 + boff);
                acc[dt] = __builtin_amdgcn_wmma_f32_16x16x32_bf16(
                    false, aP.v, false, bv[dt & 1].v, (short)0, acc[dt], false, false);
            }
        }
        __syncthreads();
    }

    // Write fp32 output [B][N][D]
#pragma unroll
    for (int dt = 0; dt < 16; ++dt) {
#pragma unroll
        for (int r = 0; r < 8; ++r) {
            out[((size_t)b * SEQ + n0 + rowCB + r) * DIM + dt * 16 + colC] =
                acc[dt][r];
        }
    }
}

// ---------------------------------------------------------------------------
extern "C" void kernel_launch(void* const* d_in, const int* in_sizes, int n_in,
                              void* d_out, int out_size, void* d_ws, size_t ws_size,
                              hipStream_t stream) {
    const float* x  = (const float*)d_in[0];
    const float* Wq = (const float*)d_in[1];
    const float* Wk = (const float*)d_in[2];
    const float* Wv = (const float*)d_in[3];
    float* out = (float*)d_out;

    char* ws = (char*)d_ws;
    size_t off = 0;
    unsigned short* xb  = (unsigned short*)(ws + off); off += (size_t)BATCH * SEQ * DIM * 2;
    unsigned short* wqb = (unsigned short*)(ws + off); off += (size_t)DIM * DIM * 2;
    unsigned short* wkb = (unsigned short*)(ws + off); off += (size_t)DIM * DIM * 2;
    unsigned short* wvb = (unsigned short*)(ws + off); off += (size_t)DIM * DIM * 2;
    unsigned short* Qb  = (unsigned short*)(ws + off); off += (size_t)BATCH * SEQ * DIM * 2;
    unsigned short* Kb  = (unsigned short*)(ws + off); off += (size_t)BATCH * SEQ * DIM * 2;
    unsigned short* Vtb = (unsigned short*)(ws + off); off += (size_t)BATCH * SEQ * DIM * 2;
    float* colmax = (float*)(ws + off); off += (size_t)BATCH * SEQ * 4;
    float* colsum = (float*)(ws + off); off += (size_t)BATCH * SEQ * 4;
    (void)ws_size; (void)in_sizes; (void)n_in; (void)out_size;

    const int nx = BATCH * SEQ * DIM;
    attn_cvt_bf16<<<(nx + 255) / 256, 256, 0, stream>>>(x, xb, nx);
    attn_cvt_bf16<<<(DIM * DIM + 255) / 256, 256, 0, stream>>>(Wq, wqb, DIM * DIM);
    attn_cvt_bf16<<<(DIM * DIM + 255) / 256, 256, 0, stream>>>(Wk, wkb, DIM * DIM);
    attn_cvt_bf16<<<(DIM * DIM + 255) / 256, 256, 0, stream>>>(Wv, wvb, DIM * DIM);

    attn_qkv_gemm<<<BATCH * NT16, 96, 0, stream>>>(xb, wqb, wkb, wvb, Qb, Kb, Vtb);
    attn_col_stats<<<BATCH * NT16, 32, 0, stream>>>(Qb, Kb, colmax, colsum);
    attn_out_gemm<<<BATCH * NT16, 32, 0, stream>>>(Qb, Kb, Vtb, colmax, colsum, out);
}
